// SimpleGATLayer_70360154243584
// MI455X (gfx1250) — matrix operand order repaired and checked by reading, not compiled, not run on hardware
//
#include <hip/hip_runtime.h>
#include <stdint.h>

typedef __attribute__((ext_vector_type(16))) _Float16 v16h;
typedef __attribute__((ext_vector_type(8)))  float    v8f;

// generic vector type matching the async-copy builtin's expected pointee
typedef int vec4i __attribute__((vector_size(16)));
typedef __attribute__((address_space(1))) vec4i* gptr_v4i;
typedef __attribute__((address_space(3))) vec4i* lptr_v4i;

#define NN     3072
#define INDIM  128
#define HEADS  4
#define ODIM   64
#define CDIM   256   // HEADS*ODIM
#define EDIM   8
#define NEGMIN (-3.4028234663852886e38f)

// i-stride inside an LDS edge tile row: 8 data dwords + 4 pad dwords -> the 16
// lanes' ds_load_b128 addresses (i*12 dwords apart) cover all 64 banks once.
#define ISTRIDE 12
#define JSTRIDE (16 * ISTRIDE)          // 192 dwords per j-line
#define TILE_DW (32 * JSTRIDE)          // 6144 dwords = 24KB per tile buffer

// ---- CDNA5 async global->LDS copy (16B), with sync fallback -------------
__device__ __forceinline__ void async_copy16(const void* gsrc, void* ldst) {
#if __has_builtin(__builtin_amdgcn_global_load_async_to_lds_b128)
  __builtin_amdgcn_global_load_async_to_lds_b128(
      (gptr_v4i)(void*)gsrc, (lptr_v4i)ldst, 0, 0);
#else
  *(float4*)ldst = *(const float4*)gsrc;
#endif
}

__device__ __forceinline__ void wait_async0() {
#if __has_builtin(__builtin_amdgcn_s_wait_asynccnt)
  __builtin_amdgcn_s_wait_asynccnt(0);
#else
  asm volatile("s_wait_asynccnt 0" ::: "memory");
#endif
}

// ---------------------------------------------------------------- kernel 1
__global__ __launch_bounds__(256) void proj_kernel(const float* __restrict__ x,
                                                   const float* __restrict__ W,
                                                   float* __restrict__ h) {
  __shared__ float xs[INDIM];
  const int row = blockIdx.x;
  const int c = threadIdx.x;
  if (c < INDIM) xs[c] = x[row * INDIM + c];
  __syncthreads();
  float acc = 0.f;
#pragma unroll 8
  for (int k = 0; k < INDIM; ++k) acc += xs[k] * W[k * CDIM + c];
  h[row * CDIM + c] = acc;
}

// ---------------------------------------------------------------- kernel 2
__global__ void score_kernel(const float* __restrict__ h,
                             const float* __restrict__ a_src,
                             const float* __restrict__ a_dst,
                             float* __restrict__ s_src,
                             float* __restrict__ s_dst) {
  const int id = blockIdx.x * blockDim.x + threadIdx.x;
  if (id >= NN * HEADS) return;
  const int n = id >> 2, hh = id & 3;
  const float* hp = h + n * CDIM + hh * ODIM;
  const float* as = a_src + hh * ODIM;
  const float* ad = a_dst + hh * ODIM;
  float s1 = 0.f, s2 = 0.f;
#pragma unroll 8
  for (int d = 0; d < ODIM; ++d) { float v = hp[d]; s1 += v * as[d]; s2 += v * ad[d]; }
  s_src[id] = s1;
  s_dst[id] = s2;
}

// ---------------------------------------------------------------- kernel 3
// Pre-pack h into WMMA B-operand layout (f16), per head / j-tile / n-tile.
#define PACK_TOTAL (HEADS * 96 * 4 * 32 * 8)
__global__ void pack_kernel(const float* __restrict__ h, uint32_t* __restrict__ hpack) {
  int t = blockIdx.x * blockDim.x + threadIdx.x;
  if (t >= PACK_TOTAL) return;
  int t2 = t;
  const int v    = t2 & 7;  t2 >>= 3;
  const int lane = t2 & 31; t2 >>= 5;
  const int nt   = t2 & 3;  t2 >>= 2;
  const int jt   = t2 % 96;
  const int hh   = t2 / 96;
  const int ncol = lane & 15;
  const int kb   = (lane >= 16) ? 8 : 0;
  const int j    = jt * 32 + kb + 2 * v + ((v >= 4) ? 8 : 0);
  const int d    = hh * ODIM + nt * 16 + ncol;
  union { _Float16 hf[2]; uint32_t u; } pk;
  pk.hf[0] = (_Float16)h[(size_t)j * CDIM + d];
  pk.hf[1] = (_Float16)h[(size_t)(j + 1) * CDIM + d];
  hpack[t] = pk.u;
}

// ---------------------------------------------------------------- kernel 4
// Fused flash-style GAT: async-stage edge tile to LDS once per WGP, then all
// 4 head-waves compute score -> leaky -> mask -> online softmax -> P@H (WMMA).
__global__ __launch_bounds__(128) void gat_fused(const float* __restrict__ s_src,
                                                 const float* __restrict__ s_dst,
                                                 const int*   __restrict__ adj,
                                                 const float* __restrict__ edge_feats,
                                                 const float* __restrict__ ep_w,
                                                 const float* __restrict__ ep_b,
                                                 const uint32_t* __restrict__ hpack,
                                                 float* __restrict__ out_pre) {
  __shared__ __align__(16) float sef[2][TILE_DW];   // 2 x 24KB double buffer

  const int tid    = threadIdx.x;
  const int lane   = tid & 31;
  const int hh     = tid >> 5;                   // head = wave id
  const int i0     = blockIdx.x * 16;
  const int iloc   = lane & 15;                  // A-layout row owned by this lane
  const int i      = i0 + iloc;
  const int hiHalf = (lane >= 16) ? 8 : 0;       // upper half owns K+8 / M+8

  float epw[EDIM];
#pragma unroll
  for (int k = 0; k < EDIM; ++k) epw[k] = ep_w[k * HEADS + hh];
  const float epb    = ep_b[hh];
  const float ssrc_i = s_src[i * HEADS + hh];

  const int* adj_row = adj + (size_t)i * NN;

  // stage one 16x32 edge tile (1024 x 16B chunks, 8 per thread) into LDS buf
  auto stage = [&](int buf, int jt) {
    const int j0 = jt * 32;
#pragma unroll
    for (int q = 0; q < 8; ++q) {
      const int c     = tid * 8 + q;
      const int j_loc = c >> 5;
      const int r     = c & 31;
      const int i_loc = r >> 1;
      const int half  = r & 1;
      const float* gsrc = edge_feats +
          ((size_t)(i0 + i_loc) * NN + (j0 + j_loc)) * EDIM + half * 4;
      float* ldst = &sef[buf][j_loc * JSTRIDE + i_loc * ISTRIDE + half * 4];
      async_copy16(gsrc, ldst);
    }
  };

  v8f acc0 = {}, acc1 = {}, acc2 = {}, acc3 = {};
  float m = NEGMIN, l = 0.f;

  stage(0, 0);
  wait_async0();
  __syncthreads();

  for (int jt = 0; jt < 96; ++jt) {
    const int buf = jt & 1;
    if (jt + 1 < 96) stage(buf ^ 1, jt + 1);   // overlap next tile with compute

    const int jg = jt * 32;
    float p[16];
    float tmax = NEGMIN;
    // scores in WMMA A-matrix element order:
    // j_loc = hiHalf + (e<8 ? e : e+8)
#pragma unroll
    for (int e = 0; e < 16; ++e) {
      const int j_loc = hiHalf + ((e < 8) ? e : e + 8);
      const int j     = jg + j_loc;
      const float* efp = &sef[buf][j_loc * JSTRIDE + iloc * ISTRIDE];
      const float4 e0 = *(const float4*)(efp);
      const float4 e1 = *(const float4*)(efp + 4);
      float ea = epb + e0.x * epw[0] + e0.y * epw[1] + e0.z * epw[2] + e0.w * epw[3]
                     + e1.x * epw[4] + e1.y * epw[5] + e1.z * epw[6] + e1.w * epw[7];
      float s = ssrc_i + s_dst[j * HEADS + hh] + ea;
      s = (s > 0.f) ? s : 0.2f * s;            // leaky_relu(0.2)
      s = (adj_row[j] == 0) ? NEGMIN : s;      // mask after leaky (as reference)
      p[e] = s;
      tmax = fmaxf(tmax, s);
    }
    tmax = fmaxf(tmax, __shfl_xor(tmax, 16, 32));
    const float mnew = fmaxf(m, tmax);
    float tsum = 0.f;
    v16h a;
#pragma unroll
    for (int e = 0; e < 16; ++e) {
      const float pe = __expf(p[e] - mnew);
      tsum += pe;
      a[e] = (_Float16)pe;
    }
    tsum += __shfl_xor(tsum, 16, 32);
    const float scale = __expf(m - mnew);
    l = l * scale + tsum;
    m = mnew;
#pragma unroll
    for (int r = 0; r < 8; ++r) {
      const float sc = __shfl(scale, r + hiHalf, 32);
      acc0[r] *= sc; acc1[r] *= sc; acc2[r] *= sc; acc3[r] *= sc;
    }
    const uint32_t* bbase = hpack + ((size_t)((hh * 96 + jt) * 4) * 32 + lane) * 8;
    const v16h b0 = *(const v16h*)(bbase);
    const v16h b1 = *(const v16h*)(bbase + 32 * 8);
    const v16h b2 = *(const v16h*)(bbase + 2 * 32 * 8);
    const v16h b3 = *(const v16h*)(bbase + 3 * 32 * 8);
    acc0 = __builtin_amdgcn_wmma_f32_16x16x32_f16(false, a, false, b0, (short)0, acc0, false, false);
    acc1 = __builtin_amdgcn_wmma_f32_16x16x32_f16(false, a, false, b1, (short)0, acc1, false, false);
    acc2 = __builtin_amdgcn_wmma_f32_16x16x32_f16(false, a, false, b2, (short)0, acc2, false, false);
    acc3 = __builtin_amdgcn_wmma_f32_16x16x32_f16(false, a, false, b3, (short)0, acc3, false, false);

    wait_async0();      // next tile landed
    __syncthreads();    // all waves done with current buf
  }

  // normalize and store: one reciprocal per lane, broadcast per C-row
  const float linv = 1.f / l;
#pragma unroll
  for (int r = 0; r < 8; ++r) {
    const float inv = __shfl(linv, r + hiHalf, 32);
    const int   row = i0 + r + hiHalf;
    float* op = out_pre + (size_t)row * CDIM + hh * ODIM + (lane & 15);
    op[0]  = acc0[r] * inv;
    op[16] = acc1[r] * inv;
    op[32] = acc2[r] * inv;
    op[48] = acc3[r] * inv;
  }
}

// ---------------------------------------------------------------- kernel 5
__global__ __launch_bounds__(256) void ln_kernel(const float* __restrict__ out_pre,
                                                 const float* __restrict__ gamma,
                                                 const float* __restrict__ beta,
                                                 float* __restrict__ out) {
  __shared__ float red[CDIM];
  const int row = blockIdx.x, c = threadIdx.x;
  const float v = out_pre[(size_t)row * CDIM + c];
  red[c] = v;
  __syncthreads();
  for (int s = 128; s > 0; s >>= 1) { if (c < s) red[c] += red[c + s]; __syncthreads(); }
  const float mean = red[0] / (float)CDIM;
  __syncthreads();
  const float dv = v - mean;
  red[c] = dv * dv;
  __syncthreads();
  for (int s = 128; s > 0; s >>= 1) { if (c < s) red[c] += red[c + s]; __syncthreads(); }
  const float var = red[0] / (float)CDIM;
  out[(size_t)row * CDIM + c] = dv * rsqrtf(var + 1e-5f) * gamma[c] + beta[c];
}

// ---------------------------------------------------------------- launch
extern "C" void kernel_launch(void* const* d_in, const int* in_sizes, int n_in,
                              void* d_out, int out_size, void* d_ws, size_t ws_size,
                              hipStream_t stream) {
  const float* x          = (const float*)d_in[0];
  const int*   adj        = (const int*)  d_in[1];
  const float* edge_feats = (const float*)d_in[2];
  const float* W          = (const float*)d_in[3];
  const float* a_src      = (const float*)d_in[4];
  const float* a_dst      = (const float*)d_in[5];
  const float* ep_w       = (const float*)d_in[6];
  const float* ep_b       = (const float*)d_in[7];
  const float* gamma      = (const float*)d_in[8];
  const float* beta       = (const float*)d_in[9];
  float* out = (float*)d_out;

  // workspace carve-up (all offsets 256B-aligned); ~7.6 MB total
  char* ws = (char*)d_ws;
  float*    h       = (float*)(ws);                    // 3072*256*4     = 3145728
  float*    s_src   = (float*)(ws + 3145728);          // 49152
  float*    s_dst   = (float*)(ws + 3194880);          // 49152
  float*    out_pre = (float*)(ws + 3244032);          // 3145728
  uint32_t* hpack   = (uint32_t*)(ws + 6389760);       // 1572864

  proj_kernel <<<NN, CDIM, 0, stream>>>(x, W, h);
  score_kernel<<<(NN * HEADS + 255) / 256, 256, 0, stream>>>(h, a_src, a_dst, s_src, s_dst);
  pack_kernel <<<PACK_TOTAL / 256, 256, 0, stream>>>(h, hpack);
  gat_fused   <<<NN / 16, 128, 0, stream>>>(s_src, s_dst, adj, edge_feats, ep_w, ep_b, hpack, out_pre);
  ln_kernel   <<<NN, CDIM, 0, stream>>>(out_pre, gamma, beta, out);
}
